// DGMGearnet_25202868093418
// MI455X (gfx1250) — matrix operand order repaired and checked by reading, not compiled, not run on hardware
//
#include <hip/hip_runtime.h>
#include <hip/hip_bf16.h>
#include <stdint.h>
#include <stddef.h>

// ---------------------------------------------------------------------------
// DGM-GearNet pipeline for gfx1250 (MI455X), wave32 + WMMA f16->f32.
// Dense-adjacency formulation; all GEMM B-operands kept transposed (NxK).
// GEMM: double-buffered async LDS staging (ASYNCcnt) + 4 WMMA per K-slab.
// NE<-AT bulk copies routed through the Tensor Data Mover (TENSORcnt).
// ---------------------------------------------------------------------------

#if defined(__has_builtin)
#  if __has_builtin(__builtin_amdgcn_tensor_load_to_lds) && \
      __has_builtin(__builtin_amdgcn_tensor_store_from_lds)
#    define HAVE_TDM 1
#  endif
#endif
#ifndef HAVE_TDM
#  define HAVE_TDM 0
#endif

namespace {

constexpr int kN   = 2048;
constexpr int kR   = 3;
constexpr int kNR  = kN * kR;       // 6144
constexpr int kD   = 128;           // D_IN = D_H = S_IN
constexpr int kSO  = 64;            // S_OUT
constexpr int kE   = 32768;
constexpr int kWin = 256;
constexpr int kK   = 16;

typedef __attribute__((ext_vector_type(16))) _Float16 v16h;
typedef __attribute__((ext_vector_type(8)))  _Float16 v8h;
typedef __attribute__((ext_vector_type(8)))  float    v8f;

__device__ __forceinline__ unsigned lds_off(const void* p) {
  return (unsigned)(unsigned long long)
      (__attribute__((address_space(3))) const char*)p;
}

// ---------------------------------------------------------------- utilities
__global__ __launch_bounds__(256)
void k_zero(float* __restrict__ p, size_t n) {
  size_t stride = (size_t)gridDim.x * blockDim.x;
  for (size_t i = (size_t)blockIdx.x * blockDim.x + threadIdx.x; i < n; i += stride)
    p[i] = 0.0f;
}

__global__ __launch_bounds__(256)
void k_cvt(const float* __restrict__ s, _Float16* __restrict__ d, size_t n) {
  size_t stride = (size_t)gridDim.x * blockDim.x;
  for (size_t i = (size_t)blockIdx.x * blockDim.x + threadIdx.x; i < n; i += stride)
    d[i] = (_Float16)s[i];
}

// dst(N x M) f16 = transpose of src(M x N) f32
__global__ __launch_bounds__(256)
void k_cvt_t(const float* __restrict__ s, _Float16* __restrict__ d, int M, int N) {
  int idx = blockIdx.x * blockDim.x + threadIdx.x;
  if (idx >= M * N) return;
  int r = idx / N, c = idx % N;
  d[(size_t)c * M + r] = (_Float16)s[idx];
}

__global__ __launch_bounds__(256)
void k_scatter_edges(const int* __restrict__ ei, const int* __restrict__ er,
                     const float* __restrict__ ew, float* __restrict__ AT) {
  int e = blockIdx.x * blockDim.x + threadIdx.x;
  if (e >= kE) return;
  int ni = ei[2 * e + 0];
  int no = ei[2 * e + 1];
  int r  = er[e];
  atomicAdd(AT + ((size_t)r * kN + no) * kN + ni, ew[e]);
}

// ------------------------------------------------------- TDM bulk copy (f32)
// One wave moves 8192 f32 (32KB) global->LDS->global via the Tensor Data
// Mover. D# group0: {count=1, lds_addr, global_addr[56:0], type=2};
// group1: data_size=4B, tensor_dim0 = tile_dim0 = 8192.
constexpr int kTdmElems = 8192;

__global__ __launch_bounds__(32)
void k_tdm_copy(const float* __restrict__ src, float* __restrict__ dst)
{
#if HAVE_TDM
  __shared__ float buf[kTdmElems];
  const size_t base = (size_t)blockIdx.x * kTdmElems;
  const unsigned long long gs = (unsigned long long)(src + base);
  const unsigned long long gd = (unsigned long long)(dst + base);
  const unsigned lb = lds_off(buf);

  typedef __attribute__((ext_vector_type(4))) unsigned v4u;
  typedef __attribute__((ext_vector_type(8))) int v8i;
  typedef __attribute__((ext_vector_type(4))) int v4i;

  const v8i g1 = { (int)(2u << 16)                               /* data_size=4B */,
                   (int)(((unsigned)kTdmElems & 0xFFFFu) << 16)  /* tdim0[15:0]  */,
                   (int)((unsigned)kTdmElems >> 16)              /* tdim0[31:16] */,
                   (int)(((unsigned)kTdmElems & 0xFFFFu) << 16)  /* tile_dim0    */,
                   0, 0, 0, 0 };
  const v4i z4 = { 0, 0, 0, 0 };
  const v8i z8 = { 0, 0, 0, 0, 0, 0, 0, 0 };

  const v4u g0l = { 1u, lb, (unsigned)gs,
                    (unsigned)((gs >> 32) & 0x1FFFFFFull) | (2u << 30) };
  __builtin_amdgcn_tensor_load_to_lds(g0l, g1, z4, z4, z8, 0);
#if __has_builtin(__builtin_amdgcn_s_wait_tensorcnt)
  __builtin_amdgcn_s_wait_tensorcnt((short)0);
#else
  asm volatile("s_wait_tensorcnt 0x0" ::: "memory");
#endif

  const v4u g0s = { 1u, lb, (unsigned)gd,
                    (unsigned)((gd >> 32) & 0x1FFFFFFull) | (2u << 30) };
  __builtin_amdgcn_tensor_store_from_lds(g0s, g1, z4, z4, z8, 0);
#if __has_builtin(__builtin_amdgcn_s_wait_tensorcnt)
  __builtin_amdgcn_s_wait_tensorcnt((short)0);
#else
  asm volatile("s_wait_tensorcnt 0x0" ::: "memory");
#endif
#else
  const size_t base = (size_t)blockIdx.x * kTdmElems;
  for (int i = threadIdx.x; i < kTdmElems; i += 32)
    dst[base + i] = src[base + i];
#endif
}

// ------------------------------------------------------------------- GEMM
// C(MxNc) = A(f16, MxKc row-major) * BT(f16, NcxKc row-major)^T
// Block: 256 threads = 8 waves (4x2); block tile 128x64; wave tile 32x32.
// Double-buffered K-slabs staged via global_load_async_to_lds_b128; each wave
// issues exactly 3 async ops per slab -> s_wait_asynccnt 0x3 overlaps compute
// with the next slab's fill.
constexpr int BM = 128, BN = 64, BK = 32;
constexpr int LDT = 40;  // LDS row pitch in halves (32 data + 8 pad = 80B)

__global__ __launch_bounds__(256)
void k_wmma_gemm(const _Float16* __restrict__ A, const _Float16* __restrict__ BT,
                 float* __restrict__ Cf, _Float16* __restrict__ Ch,
                 const float* __restrict__ bias,
                 int M, int Nc, int Kc, int relu, int transCh)
{
  __shared__ _Float16 Al[2][BM * LDT];
  __shared__ _Float16 Bl[2][BN * LDT];

  const int tid  = threadIdx.x;
  const int lane = tid & 31;
  const int wave = tid >> 5;
  const int wm = wave & 3;            // 4 wave rows  (32 M each)
  const int wn = wave >> 2;           // 2 wave cols  (32 N each)
  const int m0 = blockIdx.x * BM;
  const int n0 = blockIdx.y * BN;

  // ISA 7.12.2 16-bit fragment geometry
  const int am  = lane & 15;
  const int akb = (lane < 16) ? 0 : 8;
  const int bn  = lane & 15;
  const int bkb = (lane < 16) ? 0 : 8;

  // branch-free staging: thread -> {A row srow, A row srow+64, BT row srow}
  const int srow = tid >> 2;          // 0..63
  const int sseg = (tid & 3) * 8;     // half offset within 32-half row
  const _Float16* gA0 = A  + (size_t)(m0 + srow)      * Kc + sseg;
  const _Float16* gA1 = A  + (size_t)(m0 + srow + 64) * Kc + sseg;
  const _Float16* gB0 = BT + (size_t)(n0 + srow)      * Kc + sseg;
  const unsigned laA0[2] = { lds_off(&Al[0][srow * LDT + sseg]),
                             lds_off(&Al[1][srow * LDT + sseg]) };
  const unsigned laA1[2] = { lds_off(&Al[0][(srow + 64) * LDT + sseg]),
                             lds_off(&Al[1][(srow + 64) * LDT + sseg]) };
  const unsigned laB[2]  = { lds_off(&Bl[0][srow * LDT + sseg]),
                             lds_off(&Bl[1][srow * LDT + sseg]) };

  auto issue = [&](int k0, int buf) {
    asm volatile("global_load_async_to_lds_b128 %0, %1, off"
                 :: "v"(laA0[buf]), "v"((unsigned long long)(gA0 + k0)) : "memory");
    asm volatile("global_load_async_to_lds_b128 %0, %1, off"
                 :: "v"(laA1[buf]), "v"((unsigned long long)(gA1 + k0)) : "memory");
    asm volatile("global_load_async_to_lds_b128 %0, %1, off"
                 :: "v"(laB[buf]),  "v"((unsigned long long)(gB0 + k0)) : "memory");
  };

  v8f acc[2][2] = {};
  const int nslab = Kc / BK;

  issue(0, 0);
  for (int t = 0; t < nslab; ++t) {
    const int cur = t & 1;
    if (t + 1 < nslab) {
      issue((t + 1) * BK, cur ^ 1);
      asm volatile("s_wait_asynccnt 0x3" ::: "memory");   // slab t landed
    } else {
      asm volatile("s_wait_asynccnt 0x0" ::: "memory");
    }
    __syncthreads();

    // ---- fragments from LDS (aligned 16B ds loads)
    v16h af[2], bf[2];
#pragma unroll
    for (int mt = 0; mt < 2; ++mt) {
      const int row = wm * 32 + mt * 16 + am;
      v8h lo = *(const v8h*)&Al[cur][row * LDT + akb];
      v8h hi = *(const v8h*)&Al[cur][row * LDT + akb + 16];
      af[mt] = __builtin_shufflevector(lo, hi, 0, 1, 2, 3, 4, 5, 6, 7,
                                       8, 9, 10, 11, 12, 13, 14, 15);
    }
#pragma unroll
    for (int nt = 0; nt < 2; ++nt) {
      const int col = wn * 32 + nt * 16 + bn;
      v8h lo = *(const v8h*)&Bl[cur][col * LDT + bkb];
      v8h hi = *(const v8h*)&Bl[cur][col * LDT + bkb + 16];
      bf[nt] = __builtin_shufflevector(lo, hi, 0, 1, 2, 3, 4, 5, 6, 7,
                                       8, 9, 10, 11, 12, 13, 14, 15);
    }
#pragma unroll
    for (int mt = 0; mt < 2; ++mt)
#pragma unroll
      for (int nt = 0; nt < 2; ++nt)
        acc[mt][nt] = __builtin_amdgcn_wmma_f32_16x16x32_f16(
            false, af[mt], false, bf[nt], (short)0, acc[mt][nt], false, false);

    __syncthreads();
  }

  // ---- writeback (C layout: VGPR i -> row +i / +8+i, col = lane&15)
#pragma unroll
  for (int mt = 0; mt < 2; ++mt) {
#pragma unroll
    for (int nt = 0; nt < 2; ++nt) {
      const int col  = n0 + wn * 32 + nt * 16 + (lane & 15);
      const int row0 = m0 + wm * 32 + mt * 16 + ((lane < 16) ? 0 : 8);
      const float bv = bias ? bias[col] : 0.0f;
#pragma unroll
      for (int i = 0; i < 8; ++i) {
        float v = acc[mt][nt][i] + bv;
        if (relu) v = fmaxf(v, 0.0f);
        if (Cf) Cf[(size_t)(row0 + i) * Nc + col] = v;
        if (Ch) {
          const size_t o = transCh ? ((size_t)col * M + (row0 + i))
                                   : ((size_t)(row0 + i) * Nc + col);
          Ch[o] = (_Float16)v;
        }
      }
    }
  }
}

// -------------------------------------------------------- banded top-K rewire
__global__ __launch_bounds__(32)
void k_topk_rewire(const float* __restrict__ q, const float* __restrict__ kmat,
                   float* __restrict__ NE)
{
  const int rn   = blockIdx.x;       // r*N + n
  const int r    = rn >> 11;
  const int n    = rn & (kN - 1);
  const int lane = threadIdx.x;

  __shared__ float qs[kSO];
  __shared__ float sc[2 * kWin + 8];
  __shared__ float tv[kK];
  __shared__ int   ti[kK];

  qs[lane]      = q[(size_t)rn * kSO + lane];
  qs[lane + 32] = q[(size_t)rn * kSO + lane + 32];
  __syncthreads();

  int lo = n - kWin; if (lo < 0) lo = 0;
  int hi = n + kWin; if (hi > kN - 1) hi = kN - 1;
  const int cnt = hi - lo + 1;        // always >= 257 > 16

  for (int idx = lane; idx < cnt; idx += 32) {
    const float* kr = kmat + ((size_t)r * kN + (lo + idx)) * kSO;
    __builtin_prefetch(kr + 32 * kSO, 0, 0);
    float s = 0.0f;
#pragma unroll 8
    for (int d = 0; d < kSO; ++d) s += qs[d] * kr[d];
    sc[idx] = s * (1.0f / 16.0f);     // / (H * sqrt(dh)) = 1/16
  }
  __syncthreads();

  for (int t = 0; t < kK; ++t) {
    float best = -3.0e38f; int bi = 0x7fffffff;
    for (int idx = lane; idx < cnt; idx += 32) {
      float v = sc[idx];
      if (v > best) { best = v; bi = idx; }
    }
    for (int off = 16; off > 0; off >>= 1) {
      float ov = __shfl_xor(best, off, 32);
      int   oi = __shfl_xor(bi,   off, 32);
      if (ov > best || (ov == best && oi < bi)) { best = ov; bi = oi; }
    }
    if (lane == 0) { tv[t] = best; ti[t] = bi; sc[bi] = -3.0e38f; }
    __syncthreads();
  }

  const float vmax = tv[0];           // first pick is the max
  float e = (lane < kK) ? __expf((tv[lane] - vmax) * 2.0f) : 0.0f;  // TEMP=0.5
  float sum = e;
  for (int off = 16; off > 0; off >>= 1) sum += __shfl_xor(sum, off, 32);
  if (lane < kK) {
    const float soft = e / sum;
    const int col = lo + ti[lane];
    float* p = NE + ((size_t)r * kN + col) * kN + n;
    *p = fmaxf(*p, soft);
  }
}

// ------------------------------------------------------------- pack kernels
__global__ __launch_bounds__(256)
void k_pack_cat(const _Float16* __restrict__ agg, const _Float16* __restrict__ tailT,
                _Float16* __restrict__ cat)
{
  int idx = blockIdx.x * blockDim.x + threadIdx.x;
  if (idx >= kN * 512) return;
  int n = idx >> 9, c = idx & 511;
  _Float16 v;
  if (c < 384) {
    int r = c >> 7, d = c & 127;
    v = agg[((size_t)r * kN + n) * kD + d];
  } else {
    v = tailT[(size_t)(c - 384) * kN + n];
  }
  cat[idx] = v;
}

__global__ __launch_bounds__(256)
void k_pack_si1T(const _Float16* __restrict__ h0T, const _Float16* __restrict__ rel0,
                 _Float16* __restrict__ si1T)
{
  int idx = blockIdx.x * blockDim.x + threadIdx.x;
  if (idx >= 512 * kN) return;
  int c = idx >> 11, n = idx & (kN - 1);
  _Float16 v;
  if (c < 128) {
    v = h0T[(size_t)c * kN + n];
  } else {
    int r = (c - 128) >> 7, d = (c - 128) & 127;
    v = rel0[((size_t)r * kN + n) * kD + d];
  }
  si1T[idx] = v;
}

__global__ __launch_bounds__(256)
void k_pack_wcatT(const float* __restrict__ Wmain, const float* __restrict__ Wself,
                  _Float16* __restrict__ WT)
{
  int idx = blockIdx.x * blockDim.x + threadIdx.x;
  if (idx >= 128 * 512) return;
  int c = idx >> 9, r = idx & 511;
  float v = (r < 384) ? Wmain[r * 128 + c] : Wself[(r - 384) * 128 + c];
  WT[idx] = (_Float16)v;
}

__global__ void k_bias_add(const float* __restrict__ a, const float* __restrict__ b,
                           float* __restrict__ o) {
  int d = threadIdx.x;
  if (d < kD) o[d] = a[d] + b[d];
}

__global__ __launch_bounds__(128)
void k_colsum(const float* __restrict__ h1, float* __restrict__ out) {
  int d = threadIdx.x;
  float s = 0.0f;
  for (int n = 0; n < kN; ++n) s += h1[(size_t)n * kD + d];
  out[d] = s;
}

}  // namespace

// ---------------------------------------------------------------------------
extern "C" void kernel_launch(void* const* d_in, const int* in_sizes, int n_in,
                              void* d_out, int out_size, void* d_ws, size_t ws_size,
                              hipStream_t stream) {
  (void)in_sizes; (void)n_in; (void)out_size; (void)ws_size;

  const float* x    = (const float*)d_in[0];
  const int*   ei   = (const int*)d_in[1];
  const int*   er   = (const int*)d_in[2];
  const float* ew   = (const float*)d_in[3];
  const float* Ws0  = (const float*)d_in[4];
  const float* bs0  = (const float*)d_in[5];
  const float* Ws1  = (const float*)d_in[6];
  const float* bs1  = (const float*)d_in[7];
  const float* Wq0  = (const float*)d_in[8];
  const float* Wk0  = (const float*)d_in[9];
  const float* Wq1  = (const float*)d_in[10];
  const float* Wk1  = (const float*)d_in[11];
  const float* W0   = (const float*)d_in[12];
  const float* b0   = (const float*)d_in[13];
  const float* Wsl0 = (const float*)d_in[14];
  const float* bsl0 = (const float*)d_in[15];
  const float* W1   = (const float*)d_in[16];
  const float* b1   = (const float*)d_in[17];
  const float* Wsl1 = (const float*)d_in[18];
  const float* bsl1 = (const float*)d_in[19];

  float* out = (float*)d_out;            // [0:128] graph, [128:] node (2048x128)
  float* h1  = out + kD;

  // ----- workspace carve
  char* ws = (char*)d_ws;
  size_t off = 0;
  auto carve = [&](size_t bytes) -> void* {
    void* p = ws + off;
    off += (bytes + 255) & ~(size_t)255;
    return p;
  };
  const size_t adjN = (size_t)kR * kN * kN;          // 12.58M elems
  float*    AT    = (float*)   carve(adjN * 4);
  float*    NE    = (float*)   carve(adjN * 4);
  _Float16* ATh   = (_Float16*)carve(adjN * 2);
  _Float16* NEh   = (_Float16*)carve(adjN * 2);
  _Float16* xT    = (_Float16*)carve((size_t)kD * kN * 2);    // 128 x 2048
  _Float16* t0h   = (_Float16*)carve((size_t)kNR * kD * 2);
  _Float16* rel0h = (_Float16*)carve((size_t)kNR * kD * 2);
  _Float16* rel1h = (_Float16*)carve((size_t)kNR * kD * 2);
  float*    qb    = (float*)   carve((size_t)kNR * kSO * 4);
  float*    kb    = (float*)   carve((size_t)kNR * kSO * 4);
  _Float16* t3a   = (_Float16*)carve((size_t)kNR * kD * 2);   // agg0 (rN)
  _Float16* t4h   = (_Float16*)carve((size_t)kNR * kD * 2);   // agg1 (rN)
  _Float16* cat0h = (_Float16*)carve((size_t)kN * 512 * 2);
  _Float16* cat1h = (_Float16*)carve((size_t)kN * 512 * 2);
  _Float16* si1T  = (_Float16*)carve((size_t)512 * kN * 2);   // 512 x 2048
  _Float16* t2a   = (_Float16*)carve((size_t)kNR * 512 * 2);
  _Float16* h0T   = (_Float16*)carve((size_t)kD * kN * 2);    // 128 x 2048
  _Float16* Ws0T  = (_Float16*)carve(128 * 128 * 2);
  _Float16* Ws1T  = (_Float16*)carve(128 * 512 * 2);
  _Float16* Wq0T  = (_Float16*)carve(64 * 128 * 2);
  _Float16* Wk0T  = (_Float16*)carve(64 * 128 * 2);
  _Float16* Wq1T  = (_Float16*)carve(64 * 128 * 2);
  _Float16* Wk1T  = (_Float16*)carve(64 * 128 * 2);
  _Float16* Wc0T  = (_Float16*)carve(128 * 512 * 2);
  _Float16* Wc1T  = (_Float16*)carve(128 * 512 * 2);
  float*    bias0c = (float*)  carve(kD * 4);
  float*    bias1c = (float*)  carve(kD * 4);

  auto gblk = [](size_t n) { return (unsigned)((n + 255) / 256); };
  auto cvt  = [&](const float* s, _Float16* d, size_t n) {
    k_cvt<<<gblk(n), 256, 0, stream>>>(s, d, n);
  };
  auto cvtT = [&](const float* s, _Float16* d, int M, int N) {
    k_cvt_t<<<gblk((size_t)M * N), 256, 0, stream>>>(s, d, M, N);
  };
  auto gemm = [&](const _Float16* A, const _Float16* BT, float* Cf, _Float16* Ch,
                  const float* bias, int M, int Nc, int Kc, int relu, int tch) {
    dim3 grid(M / BM, Nc / BN);
    k_wmma_gemm<<<grid, 256, 0, stream>>>(A, BT, Cf, Ch, bias, M, Nc, Kc, relu, tch);
  };
  auto tdm_copy = [&](const float* s, float* d) {
    k_tdm_copy<<<(unsigned)(adjN / kTdmElems), 32, 0, stream>>>(s, d);
  };

  // ----- build dense transposed adjacency AT[r][dst][src]
  k_zero<<<gblk(adjN), 256, 0, stream>>>(AT, adjN);
  k_scatter_edges<<<(kE + 255) / 256, 256, 0, stream>>>(ei, er, ew, AT);
  cvt(AT, ATh, adjN);

  // ----- f16 operand prep (all B operands transposed: N x K)
  cvtT(x,   xT,   kN, kD);      // 128 x 2048
  cvtT(Ws0, Ws0T, 128, 128);
  cvtT(Ws1, Ws1T, 512, 128);    // 128 x 512
  cvtT(Wq0, Wq0T, 128, 64);     // 64 x 128
  cvtT(Wk0, Wk0T, 128, 64);
  cvtT(Wq1, Wq1T, 128, 64);
  cvtT(Wk1, Wk1T, 128, 64);
  k_pack_wcatT<<<gblk(128 * 512), 256, 0, stream>>>(W0, Wsl0, Wc0T);
  k_pack_wcatT<<<gblk(128 * 512), 256, 0, stream>>>(W1, Wsl1, Wc1T);
  k_bias_add<<<1, 128, 0, stream>>>(b0, bsl0, bias0c);
  k_bias_add<<<1, 128, 0, stream>>>(b1, bsl1, bias1c);

  // ----- layer 0 sequence layer: rel0 = relu((A^T x) Ws0 + bs0)   [rN layout]
  gemm(ATh, xT, nullptr, t0h, nullptr, kNR, kD, kN, 0, 0);
  gemm(t0h, Ws0T, nullptr, rel0h, bs0, kNR, kD, kD, 1, 0);

  // ----- rewire 0: q/k projections, banded top-K, NE0 = max(A, rewire)
  gemm(rel0h, Wq0T, qb, nullptr, nullptr, kNR, kSO, kD, 0, 0);
  gemm(rel0h, Wk0T, kb, nullptr, nullptr, kNR, kSO, kD, 0, 0);
  tdm_copy(AT, NE);
  k_topk_rewire<<<kNR, 32, 0, stream>>>(qb, kb, NE);
  cvt(NE, NEh, adjN);

  // ----- h0 = relu([ne0^T x ; x] @ [W0;Wsl0] + b0 + bsl0)   -> h0T (128x2048)
  gemm(NEh, xT, nullptr, t3a, nullptr, kNR, kD, kN, 0, 0);   // agg0 (rN layout)
  k_pack_cat<<<gblk((size_t)kN * 512), 256, 0, stream>>>(t3a, xT, cat0h);
  gemm(cat0h, Wc0T, nullptr, h0T, bias0c, kN, kD, 512, 1, 1);

  // ----- layer 1 sequence layer: rel1 = relu((ne0^T [h0;rel0]) Ws1 + bs1)
  k_pack_si1T<<<gblk((size_t)512 * kN), 256, 0, stream>>>(h0T, rel0h, si1T);
  gemm(NEh, si1T, nullptr, t2a, nullptr, kNR, 512, kN, 0, 0);
  gemm(t2a, Ws1T, nullptr, rel1h, bs1, kNR, kD, 512, 1, 0);

  // ----- rewire 1
  gemm(rel1h, Wq1T, qb, nullptr, nullptr, kNR, kSO, kD, 0, 0);
  gemm(rel1h, Wk1T, kb, nullptr, nullptr, kNR, kSO, kD, 0, 0);
  tdm_copy(AT, NE);
  k_topk_rewire<<<kNR, 32, 0, stream>>>(qb, kb, NE);
  cvt(NE, NEh, adjN);

  // ----- h1 = relu([ne1^T h0 ; h0] @ [W1;Wsl1] + b1 + bsl1)  -> d_out node feats
  gemm(NEh, h0T, nullptr, t4h, nullptr, kNR, kD, kN, 0, 0);  // agg1 (rN layout)
  k_pack_cat<<<gblk((size_t)kN * 512), 256, 0, stream>>>(t4h, h0T, cat1h);
  gemm(cat1h, Wc1T, h1, nullptr, bias1c, kN, kD, 512, 1, 0);

  // ----- graph feature = column sum of h1
  k_colsum<<<1, 128, 0, stream>>>(h1, out);
}
